// DiscriminatorMLPPremium_31250182046446
// MI455X (gfx1250) — matrix operand order repaired and checked by reading, not compiled, not run on hardware
//
#include <hip/hip_runtime.h>
#include <hip/hip_bf16.h>

// ---------------------------------------------------------------------------
// Types for CDNA5 WMMA (wave32)
// ---------------------------------------------------------------------------
typedef __attribute__((ext_vector_type(16))) __bf16 v16bf;
typedef __attribute__((ext_vector_type(8)))  float  v8f;
typedef int i32x4_t __attribute__((vector_size(16)));

__device__ __forceinline__ __bf16 f2bf(float f) {
  // round-to-nearest-even f32 -> bf16
  unsigned u = __float_as_uint(f);
  unsigned r = (u + 0x7FFFu + ((u >> 16) & 1u)) >> 16;
  unsigned short s = (unsigned short)r;
  return __builtin_bit_cast(__bf16, s);
}

// ---------------------------------------------------------------------------
// gfx1250 async global->LDS copy (16B per lane), tracked by ASYNCcnt.
// Builtin signature (from toolchain): (v4i addrspace(1)*, v4i addrspace(3)*,
// imm offset, imm cpol).  Inline-asm fallback uses the documented mnemonic.
// ---------------------------------------------------------------------------
__device__ __forceinline__ void async_cp_b128(const __bf16* g, __bf16* l) {
#if __has_builtin(__builtin_amdgcn_global_load_async_to_lds_b128)
  __builtin_amdgcn_global_load_async_to_lds_b128(
      (__attribute__((address_space(1))) i32x4_t*)(g),
      (__attribute__((address_space(3))) i32x4_t*)(l), 0, 0);
#else
  unsigned loff =
      (unsigned)(unsigned long long)(__attribute__((address_space(3))) void*)(l);
  unsigned long long ga = (unsigned long long)g;
  asm volatile("global_load_async_to_lds_b128 %0, %1, off" ::"v"(loff), "v"(ga)
               : "memory");
#endif
}

__device__ __forceinline__ void wait_async0() {
#if __has_builtin(__builtin_amdgcn_s_wait_asynccnt)
  __builtin_amdgcn_s_wait_asynccnt(0);
#else
  asm volatile("s_wait_asynccnt 0" ::: "memory");
#endif
}

// ---------------------------------------------------------------------------
// fp32 -> bf16 straight convert (activations)
// ---------------------------------------------------------------------------
__global__ void cvt_f32_bf16(const float* __restrict__ in,
                             __bf16* __restrict__ out, int n) {
  int i = blockIdx.x * blockDim.x + threadIdx.x;
  if (i < n) out[i] = f2bf(in[i]);
}

// ---------------------------------------------------------------------------
// fp32 [K][N] -> bf16 transposed [N][K]  (weights; coalesced 2-byte writes)
// ---------------------------------------------------------------------------
__global__ void cvt_transpose(const float* __restrict__ W,
                              __bf16* __restrict__ Wt, int K, int N) {
  int i = blockIdx.x * blockDim.x + threadIdx.x;  // i = n*K + k
  if (i >= K * N) return;
  int n = i / K;
  int k = i - n * K;
  Wt[i] = f2bf(W[(size_t)k * N + n]);
}

// ---------------------------------------------------------------------------
// bf16 WMMA GEMM:  C[M][N] = act( A[M][K] @ W[K][N] + bias ), W given as
// Bt[N][K].  Block tile 128x128x32, 8 waves of 64x32, double-buffered LDS
// filled by async global->LDS copies (ASYNCcnt), one barrier per K-step.
// Per K-step each wave issues 8x v_wmma_f32_16x16x32_bf16.
// M % 128 == 0, N % 128 == 0, K % 32 == 0 (true for all layers here).
// ---------------------------------------------------------------------------
template <bool BIAS, bool RELU, bool WBF, bool WF32>
__global__ __launch_bounds__(256) void gemm_bf16_wmma(
    const __bf16* __restrict__ A, const __bf16* __restrict__ Bt,
    const float* __restrict__ bias, __bf16* __restrict__ Cbf,
    float* __restrict__ Cf, int M, int N, int K) {
  constexpr int BK = 32;
  constexpr int LDK = 40;  // padded row (80B) -> conflict-free b128 frag loads
  __shared__ __bf16 sA[2][128 * LDK];
  __shared__ __bf16 sB[2][128 * LDK];

  const int tid  = threadIdx.x;
  const int lane = tid & 31;
  const int wave = tid >> 5;
  const int wm = (wave & 1) * 64;   // wave M offset in tile
  const int wn = (wave >> 1) * 32;  // wave N offset in tile
  const int bm = blockIdx.y * 128;
  const int bn = blockIdx.x * 128;

  // global tile load mapping: 256 threads x 2 rows x 16B
  const int gr = tid >> 2;        // row 0..63 (and +64)
  const int gc = (tid & 3) * 8;   // K element offset 0,8,16,24

  const __bf16* Ag = A + (size_t)(bm + gr) * K + gc;
  const __bf16* Bg = Bt + (size_t)(bn + gr) * K + gc;
  const size_t rowskip = (size_t)64 * K;

  v8f acc[4][2];
#pragma unroll
  for (int i = 0; i < 4; ++i)
#pragma unroll
    for (int j = 0; j < 2; ++j)
#pragma unroll
      for (int r = 0; r < 8; ++r) acc[i][j][r] = 0.0f;

  auto issue = [&](int kt, int buf) {
    const __bf16* a = Ag + (size_t)kt * BK;
    const __bf16* b = Bg + (size_t)kt * BK;
    async_cp_b128(a,           &sA[buf][gr * LDK + gc]);
    async_cp_b128(a + rowskip, &sA[buf][(gr + 64) * LDK + gc]);
    async_cp_b128(b,           &sB[buf][gr * LDK + gc]);
    async_cp_b128(b + rowskip, &sB[buf][(gr + 64) * LDK + gc]);
  };

  const int l  = lane & 15;
  const int hi = lane >> 4;

  auto compute = [&](int buf) {
    v16bf bfrag[2];
#pragma unroll
    for (int j = 0; j < 2; ++j) {
      // B frag (32x16): lanes 0-15: K 0..7 / 16..23 ; lanes 16-31: +8
      const __bf16* pb = &sB[buf][(wn + j * 16 + l) * LDK + hi * 8];
      union { uint4 u[2]; v16bf v; } r;
      r.u[0] = *(const uint4*)pb;
      r.u[1] = *(const uint4*)(pb + 16);
      bfrag[j] = r.v;
    }
#pragma unroll
    for (int i = 0; i < 4; ++i) {
      // A frag (16x32): lanes 0-15: K 0..7 / 16..23 ; lanes 16-31: +8
      const __bf16* pa = &sA[buf][(wm + i * 16 + l) * LDK + hi * 8];
      union { uint4 u[2]; v16bf v; } r;
      r.u[0] = *(const uint4*)pa;
      r.u[1] = *(const uint4*)(pa + 16);
      v16bf afrag = r.v;
#pragma unroll
      for (int j = 0; j < 2; ++j)
        acc[i][j] = __builtin_amdgcn_wmma_f32_16x16x32_bf16(
            false, afrag, false, bfrag[j], (short)0, acc[i][j], false, false);
    }
  };

  const int nk = K / BK;
  issue(0, 0);
  for (int kt = 0; kt < nk; ++kt) {
    wait_async0();       // own async writes for buf (kt&1) complete
    __syncthreads();     // -> every wave's writes complete / prior reads done
    if (kt + 1 < nk) issue(kt + 1, (kt + 1) & 1);
    compute(kt & 1);
  }

  // Epilogue. C/D layout: VGPR r, lanes 0-15 -> M=r, N=lane; lanes 16-31 ->
  // M=r+8, N=lane-16.
#pragma unroll
  for (int j = 0; j < 2; ++j) {
    const int col = bn + wn + j * 16 + l;
    float bv = BIAS ? bias[col] : 0.0f;
#pragma unroll
    for (int i = 0; i < 4; ++i) {
#pragma unroll
      for (int r = 0; r < 8; ++r) {
        const int row = bm + wm + i * 16 + hi * 8 + r;
        float v = acc[i][j][r] + bv;
        if (RELU) v = v > 0.0f ? v : 0.0f;
        if (WF32) Cf[(size_t)row * N + col] = v;
        if (WBF)  Cbf[(size_t)row * N + col] = f2bf(v);
      }
    }
  }
}

// ---------------------------------------------------------------------------
// Minibatch discrimination.  Ms: [2048][128*16] fp32 (col = b*16+c).
// out[i,b] = sum_j exp(-sum_c |Ms[i,b,c]-Ms[j,b,c]|).
// grid = (B=128, N/256), block = 256; j staged 128 rows at a time in LDS.
// ---------------------------------------------------------------------------
__global__ __launch_bounds__(256) void minibatch_disc(
    const float* __restrict__ Ms, float* __restrict__ outT) {
  __shared__ float mj[128][16];
  const int b = blockIdx.x;
  const int i = blockIdx.y * 256 + threadIdx.x;

  float mi[16];
  const float* pi = Ms + (size_t)i * 2048 + b * 16;
#pragma unroll
  for (int c = 0; c < 16; ++c) mi[c] = pi[c];

  float acc = 0.0f;
  for (int j0 = 0; j0 < 2048; j0 += 128) {
    __syncthreads();
    {
      const int jj = threadIdx.x >> 1;
      const int half = (threadIdx.x & 1) * 8;
      const float* pj = Ms + (size_t)(j0 + jj) * 2048 + b * 16 + half;
      *(float4*)&mj[jj][half]     = *(const float4*)pj;
      *(float4*)&mj[jj][half + 4] = *(const float4*)(pj + 4);
    }
    __syncthreads();
#pragma unroll 4
    for (int jj = 0; jj < 128; ++jj) {
      float d = 0.0f;
#pragma unroll
      for (int c = 0; c < 16; ++c) d += fabsf(mi[c] - mj[jj][c]);
      acc += __expf(-d);
    }
  }
  outT[(size_t)i * 128 + b] = acc;
}

// ---------------------------------------------------------------------------
// logits = sigmoid(concat(feature, outT) @ Wo + bo)   -> d_out[2048*512 + i]
// ---------------------------------------------------------------------------
__global__ void logits_sigmoid(const float* __restrict__ feat,
                               const float* __restrict__ outT,
                               const float* __restrict__ Wo,
                               const float* __restrict__ bo,
                               float* __restrict__ out) {
  int i = blockIdx.x * blockDim.x + threadIdx.x;
  if (i >= 2048) return;
  float s = bo[0];
  const float* f = feat + (size_t)i * 512;
#pragma unroll 8
  for (int h = 0; h < 512; ++h) s += f[h] * Wo[h];
  const float* t = outT + (size_t)i * 128;
#pragma unroll 8
  for (int b = 0; b < 128; ++b) s += t[b] * Wo[512 + b];
  out[2048 * 512 + i] = 1.0f / (1.0f + __expf(-s));
}

// ---------------------------------------------------------------------------
// launcher
// ---------------------------------------------------------------------------
extern "C" void kernel_launch(void* const* d_in, const int* in_sizes, int n_in,
                              void* d_out, int out_size, void* d_ws,
                              size_t ws_size, hipStream_t stream) {
  (void)in_sizes; (void)n_in; (void)out_size; (void)ws_size;
  const float* input = (const float*)d_in[0];
  const float* W1 = (const float*)d_in[1];
  const float* b1 = (const float*)d_in[2];
  const float* W2 = (const float*)d_in[3];
  const float* b2 = (const float*)d_in[4];
  const float* Wh = (const float*)d_in[5];
  const float* bh = (const float*)d_in[6];
  const float* W3 = (const float*)d_in[7];
  const float* b3 = (const float*)d_in[8];
  const float* W4 = (const float*)d_in[9];
  const float* b4 = (const float*)d_in[10];
  const float* Wo = (const float*)d_in[11];
  const float* bo = (const float*)d_in[12];
  const float* T  = (const float*)d_in[13];

  char* ws = (char*)d_ws;
  size_t off = 0;
  auto alloc = [&](size_t bytes) {
    char* p = ws + off;
    off += (bytes + 255) & ~(size_t)255;
    return p;
  };
  __bf16* x0     = (__bf16*)alloc((size_t)2048 * 512 * 2);
  __bf16* x1     = (__bf16*)alloc((size_t)2048 * 1024 * 2);
  __bf16* x2     = (__bf16*)alloc((size_t)2048 * 1536 * 2);
  __bf16* x3     = (__bf16*)alloc((size_t)2048 * 1536 * 2);
  __bf16* x4     = (__bf16*)alloc((size_t)2048 * 1024 * 2);
  __bf16* featbf = (__bf16*)alloc((size_t)2048 * 512 * 2);
  __bf16* W1t    = (__bf16*)alloc((size_t)512 * 1024 * 2);
  __bf16* W2t    = (__bf16*)alloc((size_t)1024 * 1536 * 2);
  __bf16* Wht    = (__bf16*)alloc((size_t)1536 * 1536 * 2);
  __bf16* W3t    = (__bf16*)alloc((size_t)1536 * 1024 * 2);
  __bf16* W4t    = (__bf16*)alloc((size_t)1024 * 512 * 2);
  __bf16* Tt     = (__bf16*)alloc((size_t)512 * 2048 * 2);
  float*  Ms     = (float*)alloc((size_t)2048 * 2048 * 4);
  float*  outT   = (float*)alloc((size_t)2048 * 128 * 4);

  float* feat_f32 = (float*)d_out;  // first output: [2048,512]

  // precision conversion (weights transposed to [N][K] bf16)
  cvt_f32_bf16<<<(2048 * 512) / 256, 256, 0, stream>>>(input, x0, 2048 * 512);
  cvt_transpose<<<(512 * 1024) / 256, 256, 0, stream>>>(W1, W1t, 512, 1024);
  cvt_transpose<<<(1024 * 1536) / 256, 256, 0, stream>>>(W2, W2t, 1024, 1536);
  cvt_transpose<<<(1536 * 1536) / 256, 256, 0, stream>>>(Wh, Wht, 1536, 1536);
  cvt_transpose<<<(1536 * 1024) / 256, 256, 0, stream>>>(W3, W3t, 1536, 1024);
  cvt_transpose<<<(1024 * 512) / 256, 256, 0, stream>>>(W4, W4t, 1024, 512);
  cvt_transpose<<<(512 * 2048) / 256, 256, 0, stream>>>(T, Tt, 512, 2048);

  // MLP stack (bf16 WMMA, fused bias+ReLU)
  gemm_bf16_wmma<true, true, true, false><<<dim3(8, 16), 256, 0, stream>>>(
      x0, W1t, b1, x1, nullptr, 2048, 1024, 512);
  gemm_bf16_wmma<true, true, true, false><<<dim3(12, 16), 256, 0, stream>>>(
      x1, W2t, b2, x2, nullptr, 2048, 1536, 1024);
  gemm_bf16_wmma<true, true, true, false><<<dim3(12, 16), 256, 0, stream>>>(
      x2, Wht, bh, x3, nullptr, 2048, 1536, 1536);
  gemm_bf16_wmma<true, true, true, false><<<dim3(8, 16), 256, 0, stream>>>(
      x3, W3t, b3, x4, nullptr, 2048, 1024, 1536);
  // feature layer: fp32 to d_out, bf16 copy for the T projection
  gemm_bf16_wmma<true, true, true, true><<<dim3(4, 16), 256, 0, stream>>>(
      x4, W4t, b4, featbf, feat_f32, 2048, 512, 1024);
  // Ms = feature @ T (no bias / no relu, fp32 out)
  gemm_bf16_wmma<false, false, false, true><<<dim3(16, 16), 256, 0, stream>>>(
      featbf, Tt, nullptr, nullptr, Ms, 2048, 2048, 512);

  // minibatch discrimination + final logits
  minibatch_disc<<<dim3(128, 8), 256, 0, stream>>>(Ms, outT);
  logits_sigmoid<<<8, 256, 0, stream>>>(feat_f32, outT, Wo, bo,
                                        (float*)d_out);
}
